// SlidingWindowAttention_22746146799913
// MI455X (gfx1250) — matrix operand order repaired and checked by reading, not compiled, not run on hardware
//
#include <hip/hip_runtime.h>
#include <stdint.h>

// ---------------------------------------------------------------------------
// Sliding-window flash attention for MI455X (gfx1250, wave32, WMMA + TDM).
// B=2 H=16 L=4096 D=128, WINDOW=1024, causal, block-windowed per reference.
//
// One 256-thread block (8 waves) computes a 128-row query tile for one (b,h).
// Each wave owns 16 query rows. Matmuls use v_wmma_f32_16x16x32_f16 with f32
// accumulation; softmax statistics in f32 (flash-attention online softmax).
// K/V tiles are DMA'd global->LDS by the Tensor Data Mover (double-buffered,
// tracked with TENSORcnt) so the copy of tile i+1 overlaps compute of tile i.
// ---------------------------------------------------------------------------

typedef _Float16 v16h __attribute__((ext_vector_type(16)));
typedef _Float16 h8   __attribute__((ext_vector_type(8)));
typedef float    v8f  __attribute__((ext_vector_type(8)));
typedef unsigned int u32x4 __attribute__((ext_vector_type(4)));
typedef int          i32x8 __attribute__((ext_vector_type(8)));
typedef int          i32x4 __attribute__((ext_vector_type(4)));

constexpr int B_  = 2;
constexpr int H_  = 16;
constexpr int L_  = 4096;
constexpr int D_  = 128;
constexpr int HALFW = 512;          // WINDOW_SIZE / 2
constexpr int QT  = 128;            // query rows per block
constexpr int KT  = 32;             // keys per inner iteration (WMMA K-dim)
constexpr int QS  = 136;            // sQ row stride (halfs), padded vs bank conflicts
constexpr int KS  = 136;            // sK row stride (halfs)
constexpr int VS  = 40;             // sVt row stride (halfs)  [keys + pad]
constexpr int PS  = 40;             // sP  row stride (halfs)  [keys + pad]

// softmax scale folded with log2(e):  exp(x*scale) == exp2(x*CEXP)
constexpr float CEXP = (float)(1.4426950408889634 / 11.313708498984761); // log2e/sqrt(128)

// Build a v16h from two 16-byte LDS chunks (A/B fragment halves).
__device__ __forceinline__ v16h ld2x8(const _Float16* a, const _Float16* b) {
    h8 lo = *(const h8*)a;
    h8 hi = *(const h8*)b;
    v16h r;
#pragma unroll
    for (int i = 0; i < 8; ++i) { r[i] = lo[i]; r[i + 8] = hi[i]; }
    return r;
}

// xor-shuffle within the wave via ds_swizzle group-of-32 mode:
// offset = { xor[14:10], or[9:5], and[4:0] }.  Masks 1/2/4/8 stay inside the
// 16-lane half-wave, which is exactly one WMMA C-tile row group.
template <int M>
__device__ __forceinline__ float swzx(float v) {
    return __int_as_float(
        __builtin_amdgcn_ds_swizzle(__float_as_int(v), (M << 10) | 0x1f));
}
__device__ __forceinline__ float rowmax16(float v) {
    v = fmaxf(v, swzx<1>(v));
    v = fmaxf(v, swzx<2>(v));
    v = fmaxf(v, swzx<4>(v));
    v = fmaxf(v, swzx<8>(v));
    return v;
}
__device__ __forceinline__ float rowsum16(float v) {
    v += swzx<1>(v);
    v += swzx<2>(v);
    v += swzx<4>(v);
    v += swzx<8>(v);
    return v;
}

// Tensor Data Mover: DMA a 2D tile (KT rows x D_ f32, row stride D_) from
// global memory into LDS at ldsAddr (dense, row-major).  Tracked by TENSORcnt.
__device__ __forceinline__ void tdm_load_tile(const float* gsrc, unsigned ldsAddr) {
    const uint64_t ga = (uint64_t)(uintptr_t)gsrc;
    u32x4 g0;
    g0[0] = 1u;                                  // count=1 (valid), no gather
    g0[1] = ldsAddr;                             // lds_addr [63:32]
    g0[2] = (unsigned)ga;                        // global_addr low
    g0[3] = (unsigned)(ga >> 32) | (2u << 30);   // global_addr high | type=2
    i32x8 g1;
    g1[0] = 0x20000;                             // data_size = 4 bytes
    g1[1] = (int)((unsigned)D_ << 16);           // tensor_dim0 = 128
    g1[2] = (int)((unsigned)L_ << 16);           // tensor_dim1 = 4096
    g1[3] = (int)((unsigned)D_ << 16);           // tile_dim0 = 128
    g1[4] = KT;                                  // tile_dim1 = 32
    g1[5] = D_;                                  // tensor_dim0_stride = 128
    g1[6] = 0;
    g1[7] = 0;
    i32x4 z4 = {0, 0, 0, 0};                     // groups 2/3 unused (2D tensor)
    i32x8 z8 = {0, 0, 0, 0, 0, 0, 0, 0};
    __builtin_amdgcn_tensor_load_to_lds(g0, g1, z4, z4, z8, 0);
}

__global__ __launch_bounds__(256, 1) void swa_wmma_kernel(
    const float* __restrict__ qg, const float* __restrict__ kg,
    const float* __restrict__ vg, float* __restrict__ outg) {
    // sQ is dead once Q fragments are in registers -> reuse its space for the
    // double-buffered f32 K/V staging tiles written by the TDM.
    __shared__ union {
        _Float16 sQ[QT * QS];                    // 34.0 KB  query tile (f16)
        float    stage[2][2][KT * D_];           // 64.0 KB  2 bufs x {K,V} f32
    } u;
    __shared__ _Float16 sK[KT * KS];             //  8.5 KB  key tile   (f16)
    __shared__ _Float16 sVt[D_ * VS];            // 10.0 KB  value tile, transposed [d][key]
    __shared__ _Float16 sP[8 * 16 * PS];         // 10.0 KB  per-wave P relayout scratch

    const int tid  = threadIdx.x;
    const int wave = tid >> 5;
    const int lane = tid & 31;
    const int ln   = lane & 15;         // column / row-in-wave index
    const int half = lane >> 4;         // which half-wave

    const int bid        = blockIdx.x;
    const int bh         = bid >> 5;            // b*H + h
    const int qTileStart = (bid & 31) * QT;
    const int sBlk       = qTileStart & ~1023;  // 1024-query block start
    const int ks         = (sBlk > HALFW) ? (sBlk - HALFW) : 0;
    const int kEnd       = qTileStart + QT;     // causal cap (= block end cap too)
    const int qw         = qTileStart + wave * 16;  // first query row of this wave

    const size_t base = (size_t)bh * L_ * D_;
    const float* qb = qg + base;
    const float* kb = kg + base;
    const float* vb = vg + base;
    float*       ob = outg + base;

    // ---- stage Q tile (128 x 128 f32 -> f16 LDS), 64 floats per thread ----
    {
        const int row = tid >> 1;
        const int seg = tid & 1;
        const float4* src = (const float4*)(qb + (size_t)(qTileStart + row) * D_ + seg * 64);
        _Float16* dst = &u.sQ[row * QS + seg * 64];
#pragma unroll
        for (int i = 0; i < 16; ++i) {
            float4 f = src[i];
            dst[4 * i + 0] = (_Float16)f.x;
            dst[4 * i + 1] = (_Float16)f.y;
            dst[4 * i + 2] = (_Float16)f.z;
            dst[4 * i + 3] = (_Float16)f.w;
        }
    }
    __syncthreads();

    // ---- load persistent Q A-fragments (16x32 each, 4 chunks over D) ----
    // A order used here (contiguous k, B matches): lane ln holds row M=ln; half 0 -> K 0..15;
    // half 1 -> K 16..31.
    v16h aq[4];
#pragma unroll
    for (int c = 0; c < 4; ++c) {
        const int d0  = 32 * c + (half ? 16 : 0);
        const int row = wave * 16 + ln;
        aq[c] = ld2x8(&u.sQ[row * QS + d0], &u.sQ[row * QS + d0 + 8]);
    }
    // Barrier lowers with s_wait_dscnt 0: all Q-fragment LDS reads have landed
    // in VGPRs, so the union space is now free for the TDM staging buffers.
    __syncthreads();

    // ---- issue TDM DMA for the first K/V tile into staging buffer 0 ----
    if (wave == 0) {
        tdm_load_tile(kb + (size_t)ks * D_, (unsigned)(uintptr_t)&u.stage[0][0][0]);
        tdm_load_tile(vb + (size_t)ks * D_, (unsigned)(uintptr_t)&u.stage[0][1][0]);
    }

    // ---- flash-attention state ----
    v8f acc[8];                       // 16 x 128 f32 output accumulator (8 C-tiles)
    float m[8], l[8];
#pragma unroll
    for (int t = 0; t < 8; ++t) acc[t] = v8f{0.f, 0.f, 0.f, 0.f, 0.f, 0.f, 0.f, 0.f};
#pragma unroll
    for (int r = 0; r < 8; ++r) { m[r] = -__builtin_inff(); l[r] = 0.f; }

    _Float16* sPw = &sP[wave * 16 * PS];

    // ---- key loop: 32 keys per iteration, TDM double-buffered ----
    int cur = 0;
    for (int kt = ks; kt < kEnd; kt += KT, cur ^= 1) {
        // (A) prev compute done (sK/sVt free) and prev conversion done
        //     (stage[cur^1] free for the next TDM write).
        __syncthreads();

        if (wave == 0) {
            if (kt + KT < kEnd) {
                const int nxt = cur ^ 1;
                tdm_load_tile(kb + (size_t)(kt + KT) * D_,
                              (unsigned)(uintptr_t)&u.stage[nxt][0][0]);
                tdm_load_tile(vb + (size_t)(kt + KT) * D_,
                              (unsigned)(uintptr_t)&u.stage[nxt][1][0]);
                // TENSORcnt completes in order: <=2 outstanding means the two
                // DMAs for tile `kt` (issued last iteration) have landed.
                __builtin_amdgcn_s_wait_tensorcnt(2);
            } else {
                __builtin_amdgcn_s_wait_tensorcnt(0);
            }
        }
        // (B) stage[cur] now holds tile kt for every wave.
        __syncthreads();

        // ---- convert staged f32 tile -> f16 sK (row major) / sVt (transposed) ----
        {
            const int row = tid >> 3;     // key row 0..31
            const int seg = tid & 7;      // 16-float segment of D
            const float4* ksrc = (const float4*)(&u.stage[cur][0][row * D_ + seg * 16]);
            const float4* vsrc = (const float4*)(&u.stage[cur][1][row * D_ + seg * 16]);
            _Float16* kdst = &sK[row * KS + seg * 16];
#pragma unroll
            for (int i = 0; i < 4; ++i) {
                float4 f = ksrc[i];
                kdst[4 * i + 0] = (_Float16)f.x;
                kdst[4 * i + 1] = (_Float16)f.y;
                kdst[4 * i + 2] = (_Float16)f.z;
                kdst[4 * i + 3] = (_Float16)f.w;
            }
#pragma unroll
            for (int i = 0; i < 4; ++i) {
                float4 f = vsrc[i];
                const int d = seg * 16 + 4 * i;
                sVt[(d + 0) * VS + row] = (_Float16)f.x;
                sVt[(d + 1) * VS + row] = (_Float16)f.y;
                sVt[(d + 2) * VS + row] = (_Float16)f.z;
                sVt[(d + 3) * VS + row] = (_Float16)f.w;
            }
        }
        // (C) f16 tiles ready.
        __syncthreads();

        // ---- S = Q * K^T for two 16-key halves ----
        v8f s0 = v8f{0.f, 0.f, 0.f, 0.f, 0.f, 0.f, 0.f, 0.f};
        v8f s1 = s0;
#pragma unroll
        for (int c = 0; c < 4; ++c) {
            const int d0 = 32 * c + (half ? 16 : 0);  // B layout: half selects K 0..15 / 16..31
            {
                const int kr = ln;                    // keys kt .. kt+15
                v16h bk = ld2x8(&sK[kr * KS + d0], &sK[kr * KS + d0 + 8]);
                s0 = __builtin_amdgcn_wmma_f32_16x16x32_f16(false, aq[c], false, bk,
                                                            (short)0, s0, false, false);
            }
            {
                const int kr = 16 + ln;               // keys kt+16 .. kt+31
                v16h bk = ld2x8(&sK[kr * KS + d0], &sK[kr * KS + d0 + 8]);
                s1 = __builtin_amdgcn_wmma_f32_16x16x32_f16(false, aq[c], false, bk,
                                                            (short)0, s1, false, false);
            }
        }

        // ---- mask + online softmax (C layout: row = r + 8*half, col = ln) ----
        const bool needMask = (kt + KT - 1) > qw;
#pragma unroll
        for (int r = 0; r < 8; ++r) {
            const int qrow = qw + r + half * 8;
            if (needMask) {
                if (qrow < kt + ln)      s0[r] = -__builtin_inff();
                if (qrow < kt + 16 + ln) s1[r] = -__builtin_inff();
            }
            const float mx   = rowmax16(fmaxf(s0[r], s1[r]));
            const float mnew = fmaxf(m[r], mx);
            const float cor  = __builtin_amdgcn_exp2f((m[r] - mnew) * CEXP);
            m[r] = mnew;
            const float p0 = __builtin_amdgcn_exp2f((s0[r] - mnew) * CEXP);
            const float p1 = __builtin_amdgcn_exp2f((s1[r] - mnew) * CEXP);
            l[r] = l[r] * cor + rowsum16(p0 + p1);
#pragma unroll
            for (int t = 0; t < 8; ++t) acc[t][r] *= cor;
            // scatter P (C layout) into per-wave LDS scratch for A relayout
            sPw[(r + half * 8) * PS + ln]      = (_Float16)p0;
            sPw[(r + half * 8) * PS + 16 + ln] = (_Float16)p1;
        }

        // wave-internal ordering for the LDS P bounce (split counter, no barrier)
        asm volatile("s_wait_dscnt 0" ::: "memory");

        // ---- O += P * V  (A = 16x32 probabilities, B = 32 keys x 16 dims) ----
        const int k0 = half ? 8 : 0;
        v16h ap = ld2x8(&sPw[ln * PS + k0], &sPw[ln * PS + k0 + 16]);
        const int kv0 = half ? 8 : 0;
#pragma unroll
        for (int t = 0; t < 8; ++t) {
            const int dr = t * 16 + ln;
            v16h bv = ld2x8(&sVt[dr * VS + kv0], &sVt[dr * VS + kv0 + 16]);
            acc[t] = __builtin_amdgcn_wmma_f32_16x16x32_f16(false, ap, false, bv,
                                                            (short)0, acc[t], false, false);
        }
    }

    // ---- epilogue: O / l, write f32 output ----
    float inv[8];
#pragma unroll
    for (int r = 0; r < 8; ++r) inv[r] = 1.0f / l[r];
#pragma unroll
    for (int t = 0; t < 8; ++t) {
        const int d = t * 16 + ln;
#pragma unroll
        for (int r = 0; r < 8; ++r) {
            const int qrow = qw + r + half * 8;
            ob[(size_t)qrow * D_ + d] = acc[t][r] * inv[r];
        }
    }
}

extern "C" void kernel_launch(void* const* d_in, const int* in_sizes, int n_in,
                              void* d_out, int out_size, void* d_ws, size_t ws_size,
                              hipStream_t stream) {
    (void)in_sizes; (void)n_in; (void)out_size; (void)d_ws; (void)ws_size;
    const float* q = (const float*)d_in[0];
    const float* k = (const float*)d_in[1];
    const float* v = (const float*)d_in[2];
    float* out = (float*)d_out;

    const int blocks = B_ * H_ * (L_ / QT);  // 1024
    swa_wmma_kernel<<<dim3(blocks), dim3(256), 0, stream>>>(q, k, v, out);
}
